// GQA_63118839382399
// MI455X (gfx1250) — matrix operand order repaired and checked by reading, not compiled, not run on hardware
//
#include <hip/hip_runtime.h>
#include <hip/hip_bf16.h>

// GQA attention layer for MI455X (gfx1250, wave32, WMMA + TDM).
//   qkv = x @ qkv_w^T ; per-head RMSNorm + RoPE on q,k ; causal GQA attention ;
//   out = y @ out_w^T
// Matmuls: v_wmma_f32_16x16x32_bf16 (fp32 accumulate). K tiles in attention are
// staged by the Tensor Data Mover (tensor_load_to_lds, double-buffered, padded
// LDS pitch), overlapped with compute; V tiles are transposed manually into LDS.

#define D_MODELC 2048
#define HQ_C 32
#define HKV_C 8
#define DHEAD_C 64
#define SEQN_C 2048
#define BATCH_C 2
#define NHQKV_C (HQ_C + 2 * HKV_C)      // 48
#define QKVN_C (NHQKV_C * DHEAD_C)      // 3072
#define MTOK_C (BATCH_C * SEQN_C)       // 4096

typedef __attribute__((ext_vector_type(16))) __bf16 bf16x16;
typedef __attribute__((ext_vector_type(8)))  float  f32x8;
typedef __attribute__((ext_vector_type(4)))  unsigned int u32x4;
typedef __attribute__((ext_vector_type(8)))  int i32x8;
typedef __attribute__((ext_vector_type(4)))  int i32x4;

union FragAB { bf16x16 v; int4 q[2]; };
union Pack4  { __bf16 b[4]; unsigned long long u; };

__device__ __forceinline__ __bf16 f2bf(float f) {
  unsigned u = __float_as_uint(f);
  unsigned r = (u + 0x7FFFu + ((u >> 16) & 1u)) >> 16;   // RNE
  unsigned short s = (unsigned short)r;
  __bf16 b; __builtin_memcpy(&b, &s, 2); return b;
}

__device__ __forceinline__ f32x8 wmma_bf16(bf16x16 a, bf16x16 b, f32x8 c) {
  // D = A(16x32) * B(32x16) + C, fp32 accumulate
  return __builtin_amdgcn_wmma_f32_16x16x32_bf16(false, a, false, b,
                                                 (short)0, c, false, false);
}

// A fragment 16x32 (MxK): lane row = lr; two contiguous 8-elem chunks:
//   elems 0-7  -> K = k0 + 8*lh + e ; elems 8-15 -> K = k0 + 16 + 8*lh + (e-8)
__device__ __forceinline__ bf16x16 load_a_frag(const __bf16* base, int ld,
                                               int k0, int lr, int lh) {
  FragAB f;
  const __bf16* p = base + (size_t)lr * ld + k0 + 8 * lh;
  f.q[0] = *(const int4*)(p);
  f.q[1] = *(const int4*)(p + 16);
  return f.v;
}

// B fragment 32x16 (KxN) from an N-major [N][K] matrix: lane col = lr,
// elems 0-7 -> K = k0 + 8*lh + e ; elems 8-15 -> K = k0 + 16 + 8*lh + (e-8).
__device__ __forceinline__ bf16x16 load_b_frag(const __bf16* base, int ld,
                                               int k0, int lr, int lh) {
  FragAB f;
  const __bf16* p = base + (size_t)lr * ld + k0 + 8 * lh;
  f.q[0] = *(const int4*)(p);
  f.q[1] = *(const int4*)(p + 16);
  return f.v;
}

// ---------------------------------------------------------- TDM 64x64 K tile
// D# per CDNA5 ISA:
//   group0: count=1 | lds_addr | global_addr[56:0] | type=2
//   group1: data_size=2B, pad_enable, pad_interval=32 DW (code 4),
//           pad_amount=4 DW (code 3)  -> LDS pitch 144B = 72 bf16 (matches
//           the compute-side layout), tensor 64x64, tile 64x64, stride0=64.
//   groups 2/3 + trailing vector: zero (2-D tile).  cpol = 0.
// This toolchain exposes the 6-arg (clang-23) builtin form.
__device__ __forceinline__ void tdm_load_k_tile(const __bf16* gsrc, __bf16* ldst) {
  unsigned long long ga = (unsigned long long)(uintptr_t)gsrc;
  unsigned lds_off = (unsigned)(uintptr_t)ldst;   // low 32 bits = LDS offset
  u32x4 g0;
  g0[0] = 1u;                                      // count = 1 (valid user D#)
  g0[1] = lds_off;                                 // lds_addr (bytes)
  g0[2] = (unsigned)(ga & 0xFFFFFFFFull);          // global_addr[31:0]
  g0[3] = (unsigned)((ga >> 32) & 0x01FFFFFFull)   // global_addr[56:32]
        | (2u << 30);                              // type = 2 ("image")
  i32x8 g1;
  g1[0] = (1 << 16)      // data_size = 2 bytes
        | (1 << 20)      // pad_enable
        | (4 << 22)      // pad_interval: code 4 = 32 DWORDs (=128B row)
        | (3 << 25);     // pad_amount: code 3 = 4 DWORDs (=16B)
  g1[1] = 64 << 16;      // tensor_dim0[15:0] = 64 (bits 63:48)
  g1[2] = 64 << 16;      // tensor_dim1[15:0] = 64 (bits 95:80)
  g1[3] = 64 << 16;      // tile_dim0 = 64 (bits 127:112)
  g1[4] = 64;            // tile_dim1 = 64 (bits 143:128); tile_dim2 = 0
  g1[5] = 64;            // tensor_dim0_stride[31:0] = 64 elems
  g1[6] = 0;
  g1[7] = 0;
  i32x4 z4 = {0, 0, 0, 0};
  i32x8 z8;
#pragma unroll
  for (int i = 0; i < 8; ++i) z8[i] = 0;
  __builtin_amdgcn_tensor_load_to_lds(g0, g1, z4, z4, z8, 0);
}

// ---------------------------------------------------------------- cast kernel
__global__ __launch_bounds__(256) void cast_f32_to_bf16_k(
    const float* __restrict__ in, __bf16* __restrict__ out, int n4) {
  int i = blockIdx.x * 256 + threadIdx.x;
  if (i >= n4) return;
  float4 f = ((const float4*)in)[i];
  Pack4 p;
  p.b[0] = f2bf(f.x); p.b[1] = f2bf(f.y);
  p.b[2] = f2bf(f.z); p.b[3] = f2bf(f.w);
  ((unsigned long long*)out)[i] = p.u;
}

// --------------------------------------------------------- WMMA GEMM (X @ W^T)
// A: MxK row-major bf16, Bw: NxK row-major bf16 (i.e. W), C: MxN fp32.
// Block = 8 waves -> 64(M) x 256(N) tile; wave tile 32x64 (8 WMMA / k-step).
__global__ __launch_bounds__(256) void gemm_bf16_nt_k(
    const __bf16* __restrict__ A, const __bf16* __restrict__ Bw,
    float* __restrict__ C, int M, int N, int K) {
  const int lane = threadIdx.x & 31;
  const int wave = threadIdx.x >> 5;
  const int lr = lane & 15, lh = lane >> 4;
  const int wm = wave >> 2, wn = wave & 3;
  const int m0 = blockIdx.x * 64 + wm * 32;
  const int n0 = blockIdx.y * 256 + wn * 64;

  f32x8 acc[2][4];
#pragma unroll
  for (int i = 0; i < 2; ++i)
#pragma unroll
    for (int nf = 0; nf < 4; ++nf)
#pragma unroll
      for (int j = 0; j < 8; ++j) acc[i][nf][j] = 0.0f;

  const __bf16* arow0 = A + (size_t)m0 * K;
  const __bf16* arow1 = A + (size_t)(m0 + 16) * K;

  for (int k0 = 0; k0 < K; k0 += 32) {
    bf16x16 a0 = load_a_frag(arow0, K, k0, lr, lh);
    bf16x16 a1 = load_a_frag(arow1, K, k0, lr, lh);
    if (k0 + 32 < K) {  // near-scope prefetch of next A tile
      __builtin_prefetch(arow0 + (size_t)lr * K + k0 + 32, 0, 3);
    }
#pragma unroll
    for (int nf = 0; nf < 4; ++nf) {
      bf16x16 b = load_b_frag(Bw + (size_t)(n0 + nf * 16) * K, K, k0, lr, lh);
      acc[0][nf] = wmma_bf16(a0, b, acc[0][nf]);
      acc[1][nf] = wmma_bf16(a1, b, acc[1][nf]);
    }
  }
  // C/D layout: VGPR j -> row = j + 8*lh, col = lr
#pragma unroll
  for (int i = 0; i < 2; ++i)
#pragma unroll
    for (int nf = 0; nf < 4; ++nf)
#pragma unroll
      for (int j = 0; j < 8; ++j) {
        int row = m0 + i * 16 + j + 8 * lh;
        int col = n0 + nf * 16 + lr;
        C[(size_t)row * N + col] = acc[i][nf][j];
      }
}

// ----------------------------------------- RMSNorm + RoPE + bf16 head packing
// qkv fp32 [b][n][48][64] -> qb [b][32][n][64], kb/vb [b][8][n][64] (bf16).
// One wave per 64-elem head row; lane holds elems {l, l+32} = RoPE pair.
__global__ __launch_bounds__(256) void norm_rope_pack_k(
    const float* __restrict__ qkv, const int* __restrict__ pos,
    const float* __restrict__ qw, const float* __restrict__ kw,
    __bf16* __restrict__ qb, __bf16* __restrict__ kb, __bf16* __restrict__ vb) {
  const int lane = threadIdx.x & 31;
  const int row = blockIdx.x * 8 + (threadIdx.x >> 5);
  const int h = row % NHQKV_C;
  const int n = (row / NHQKV_C) % SEQN_C;
  const int b = row / (NHQKV_C * SEQN_C);
  const float* src = qkv + (size_t)row * DHEAD_C;
  float t1 = src[lane], t2 = src[lane + 32];
  if (h < HQ_C + HKV_C) {
    float ss = t1 * t1 + t2 * t2;
#pragma unroll
    for (int mk = 16; mk >= 1; mk >>= 1) ss += __shfl_xor(ss, mk, 32);
    float r = rsqrtf(ss * (1.0f / 64.0f) + 1e-6f);
    const float* w = (h < HQ_C) ? qw : kw;
    t1 *= r * w[lane];
    t2 *= r * w[lane + 32];
    float p = (float)pos[n];
    float freq = __powf(10000.0f, -((float)(2 * lane)) * (1.0f / 64.0f));
    float ang = p * freq;
    float cs = __cosf(ang), sn = __sinf(ang);
    float o1 = t1 * cs - t2 * sn;
    float o2 = t1 * sn + t2 * cs;
    t1 = o1; t2 = o2;
  }
  __bf16* dst;
  if (h < HQ_C)
    dst = qb + ((size_t)(b * HQ_C + h) * SEQN_C + n) * DHEAD_C;
  else if (h < HQ_C + HKV_C)
    dst = kb + ((size_t)(b * HKV_C + (h - HQ_C)) * SEQN_C + n) * DHEAD_C;
  else
    dst = vb + ((size_t)(b * HKV_C + (h - HQ_C - HKV_C)) * SEQN_C + n) * DHEAD_C;
  dst[lane] = f2bf(t1);
  dst[lane + 32] = f2bf(t2);
}

// ------------------------------------------------ causal GQA flash attention
// Block = 8 waves, each owns 16 query rows (128-row q tile per block).
// Double-buffered LDS tiles: K staged by the TDM (padded to pitch 72),
// V transposed manually; tile kt+1 is in flight while tile kt is computed.
// Online softmax in C-fragment registers; P goes through per-wave LDS
// scratch to re-shape C-layout fp32 -> A-layout bf16 for the P@V WMMAs.
__global__ __launch_bounds__(256) void attn_fwd_k(
    const __bf16* __restrict__ qb, const __bf16* __restrict__ kb,
    const __bf16* __restrict__ vb, __bf16* __restrict__ yb) {
  __shared__ __align__(16) __bf16 lds_k[2][64 * 72];
  __shared__ __align__(16) __bf16 lds_vt[2][64 * 72];
  __shared__ __align__(16) __bf16 lds_p[8 * 16 * 64];

  const int lane = threadIdx.x & 31;
  const int wave = threadIdx.x >> 5;
  const int lr = lane & 15, lh = lane >> 4;
  const int qt = blockIdx.x;   // q tile (128 rows)
  const int hq = blockIdx.y;   // 0..31
  const int b  = blockIdx.z;   // 0..1
  const int hk = hq >> 2;      // G = 4
  const __bf16* Q  = qb + (size_t)(b * HQ_C + hq) * SEQN_C * DHEAD_C;
  const __bf16* Kh = kb + (size_t)(b * HKV_C + hk) * SEQN_C * DHEAD_C;
  const __bf16* Vh = vb + (size_t)(b * HKV_C + hk) * SEQN_C * DHEAD_C;
  const int m0 = qt * 128 + wave * 16;

  bf16x16 aq0 = load_a_frag(Q + (size_t)m0 * DHEAD_C, DHEAD_C, 0, lr, lh);
  bf16x16 aq1 = load_a_frag(Q + (size_t)m0 * DHEAD_C, DHEAD_C, 32, lr, lh);

  f32x8 o[4];
  float mrow[8], lrow[8];
#pragma unroll
  for (int df = 0; df < 4; ++df)
#pragma unroll
    for (int j = 0; j < 8; ++j) o[df][j] = 0.0f;
#pragma unroll
  for (int j = 0; j < 8; ++j) { mrow[j] = -__builtin_inff(); lrow[j] = 0.0f; }

  __bf16* pw = &lds_p[wave * 16 * 64];
  const int nkt = qt * 2 + 2;  // key tiles covering the block's causal range

  // ---- prologue: stage tile 0 into buffer 0
  if (wave == 0) tdm_load_k_tile(Kh, &lds_k[0][0]);
  for (int c = threadIdx.x; c < 512; c += 256) {   // V tile 0, transposed
    int r = c >> 3, c8 = (c & 7) * 8;
    union { int4 q; __bf16 e[8]; } u;
    u.q = *(const int4*)&Vh[(size_t)r * DHEAD_C + c8];
#pragma unroll
    for (int jj = 0; jj < 8; ++jj) lds_vt[0][(c8 + jj) * 72 + r] = u.e[jj];
  }

  for (int kt = 0; kt < nkt; ++kt) {
    const int cur = kt & 1;
    if (wave == 0) __builtin_amdgcn_s_wait_tensorcnt(0);
    __syncthreads();  // tile kt resident; everyone done computing tile kt-1

    if (kt + 1 < nkt) {  // stage tile kt+1 into the other buffer (overlapped)
      if (wave == 0)
        tdm_load_k_tile(Kh + (size_t)(kt + 1) * 64 * DHEAD_C,
                        &lds_k[cur ^ 1][0]);
      for (int c = threadIdx.x; c < 512; c += 256) {
        int r = c >> 3, c8 = (c & 7) * 8;
        union { int4 q; __bf16 e[8]; } u;
        u.q = *(const int4*)&Vh[(size_t)((kt + 1) * 64 + r) * DHEAD_C + c8];
#pragma unroll
        for (int jj = 0; jj < 8; ++jj)
          lds_vt[cur ^ 1][(c8 + jj) * 72 + r] = u.e[jj];
      }
    }
    if (kt * 64 > m0 + 15) continue;  // past this wave's causal frontier

    const __bf16* kcur = &lds_k[cur][0];
    const __bf16* vcur = &lds_vt[cur][0];

    // S = Q K^T  (4 key sub-tiles of 16, K-dim 64 = 2 wmma steps)
    f32x8 sf[4];
#pragma unroll
    for (int nf = 0; nf < 4; ++nf) {
      f32x8 cc;
#pragma unroll
      for (int j = 0; j < 8; ++j) cc[j] = 0.0f;
      cc = wmma_bf16(aq0, load_b_frag(kcur + nf * 16 * 72, 72, 0, lr, lh), cc);
      cc = wmma_bf16(aq1, load_b_frag(kcur + nf * 16 * 72, 72, 32, lr, lh), cc);
      sf[nf] = cc;
    }
    // scale, causal mask, row max (rows live in 16-lane halves)
    float rmax[8];
#pragma unroll
    for (int j = 0; j < 8; ++j) rmax[j] = -__builtin_inff();
#pragma unroll
    for (int nf = 0; nf < 4; ++nf) {
      int key = kt * 64 + nf * 16 + lr;
#pragma unroll
      for (int j = 0; j < 8; ++j) {
        int qr = m0 + j + 8 * lh;
        float v = sf[nf][j] * 0.125f;
        v = (key > qr) ? -__builtin_inff() : v;
        sf[nf][j] = v;
        rmax[j] = fmaxf(rmax[j], v);
      }
    }
#pragma unroll
    for (int j = 0; j < 8; ++j)
#pragma unroll
      for (int mk = 8; mk >= 1; mk >>= 1)
        rmax[j] = fmaxf(rmax[j], __shfl_xor(rmax[j], mk, 32));

    float corr[8], rsum[8];
#pragma unroll
    for (int j = 0; j < 8; ++j) {
      float mn = fmaxf(mrow[j], rmax[j]);
      corr[j] = __expf(mrow[j] - mn);
      mrow[j] = mn;
      rsum[j] = 0.0f;
    }
#pragma unroll
    for (int nf = 0; nf < 4; ++nf)
#pragma unroll
      for (int j = 0; j < 8; ++j) {
        float p = __expf(sf[nf][j] - mrow[j]);
        sf[nf][j] = p;
        rsum[j] += p;
      }
#pragma unroll
    for (int j = 0; j < 8; ++j) {
#pragma unroll
      for (int mk = 8; mk >= 1; mk >>= 1) rsum[j] += __shfl_xor(rsum[j], mk, 32);
      lrow[j] = lrow[j] * corr[j] + rsum[j];
    }
#pragma unroll
    for (int df = 0; df < 4; ++df)
#pragma unroll
      for (int j = 0; j < 8; ++j) o[df][j] *= corr[j];

    // P: C-layout fp32 -> per-wave LDS scratch -> A-layout bf16 fragments
#pragma unroll
    for (int nf = 0; nf < 4; ++nf)
#pragma unroll
      for (int j = 0; j < 8; ++j)
        pw[(j + 8 * lh) * 64 + nf * 16 + lr] = f2bf(sf[nf][j]);

    bf16x16 ap0 = load_a_frag(pw, 64, 0, lr, lh);
    bf16x16 ap1 = load_a_frag(pw, 64, 32, lr, lh);
    // O += P V  (V^T staged in LDS so the key dim is contiguous per lane)
#pragma unroll
    for (int df = 0; df < 4; ++df) {
      o[df] = wmma_bf16(ap0, load_b_frag(vcur + df * 16 * 72, 72, 0, lr, lh), o[df]);
      o[df] = wmma_bf16(ap1, load_b_frag(vcur + df * 16 * 72, 72, 32, lr, lh), o[df]);
    }
  }
  // y[b][n][hq*64+d] (bf16) — row-major (4096 x 2048) for the out projection
#pragma unroll
  for (int j = 0; j < 8; ++j) {
    float inv = 1.0f / lrow[j];
    int row = m0 + j + 8 * lh;
#pragma unroll
    for (int df = 0; df < 4; ++df) {
      size_t idx = ((size_t)(b * SEQN_C + row)) * D_MODELC + hq * 64 + df * 16 + lr;
      yb[idx] = f2bf(o[df][j] * inv);
    }
  }
}

// ------------------------------------------------------------------ launcher
extern "C" void kernel_launch(void* const* d_in, const int* in_sizes, int n_in,
                              void* d_out, int out_size, void* d_ws, size_t ws_size,
                              hipStream_t stream) {
  (void)in_sizes; (void)n_in; (void)out_size; (void)ws_size;
  const float* x     = (const float*)d_in[0];
  // d_in[1] = mask (bool triu k=1): causal structure applied analytically
  const int*   pos   = (const int*)d_in[2];
  const float* qkv_w = (const float*)d_in[3];
  const float* out_w = (const float*)d_in[4];
  const float* qnw   = (const float*)d_in[5];
  const float* knw   = (const float*)d_in[6];
  float* out = (float*)d_out;

  char* ws = (char*)d_ws;
  size_t off = 0;
  auto alloc = [&](size_t bytes) {
    char* p = ws + off;
    off = (off + bytes + 255) & ~(size_t)255;
    return p;
  };
  __bf16* xb   = (__bf16*)alloc((size_t)MTOK_C * D_MODELC * 2);
  __bf16* wqb  = (__bf16*)alloc((size_t)QKVN_C * D_MODELC * 2);
  __bf16* wob  = (__bf16*)alloc((size_t)D_MODELC * D_MODELC * 2);
  float*  qkvf = (float*) alloc((size_t)MTOK_C * QKVN_C * 4);
  __bf16* qbuf = (__bf16*)alloc((size_t)BATCH_C * HQ_C * SEQN_C * DHEAD_C * 2);
  __bf16* kbuf = (__bf16*)alloc((size_t)BATCH_C * HKV_C * SEQN_C * DHEAD_C * 2);
  __bf16* vbuf = (__bf16*)alloc((size_t)BATCH_C * HKV_C * SEQN_C * DHEAD_C * 2);
  __bf16* ybuf = (__bf16*)alloc((size_t)MTOK_C * D_MODELC * 2);

  int n4;
  n4 = MTOK_C * D_MODELC / 4;
  cast_f32_to_bf16_k<<<(n4 + 255) / 256, 256, 0, stream>>>(x, xb, n4);
  n4 = QKVN_C * D_MODELC / 4;
  cast_f32_to_bf16_k<<<(n4 + 255) / 256, 256, 0, stream>>>(qkv_w, wqb, n4);
  n4 = D_MODELC * D_MODELC / 4;
  cast_f32_to_bf16_k<<<(n4 + 255) / 256, 256, 0, stream>>>(out_w, wob, n4);

  gemm_bf16_nt_k<<<dim3(MTOK_C / 64, QKVN_C / 256), 256, 0, stream>>>(
      xb, wqb, qkvf, MTOK_C, QKVN_C, D_MODELC);

  norm_rope_pack_k<<<(BATCH_C * SEQN_C * NHQKV_C) / 8, 256, 0, stream>>>(
      qkvf, pos, qnw, knw, qbuf, kbuf, vbuf);

  attn_fwd_k<<<dim3(SEQN_C / 128, HQ_C, BATCH_C), 256, 0, stream>>>(
      qbuf, kbuf, vbuf, ybuf);

  gemm_bf16_nt_k<<<dim3(MTOK_C / 64, D_MODELC / 256), 256, 0, stream>>>(
      ybuf, wob, out, MTOK_C, D_MODELC, D_MODELC);
}
